// SRWKVRouter_9234179687042
// MI455X (gfx1250) — matrix operand order repaired and checked by reading, not compiled, not run on hardware
//
#include <hip/hip_runtime.h>
#include <hip/hip_bf16.h>

typedef __attribute__((ext_vector_type(8)))  float  v8f;
typedef __attribute__((ext_vector_type(16))) __bf16 v16bf;

#define D_DIM     2048
#define E_DIM     64
#define TOPK      8
#define ROWS_PER_BLOCK 128
#define KITERS    (D_DIM / 32)          // 64
#define B_UINTS   (KITERS * 4 * 32 * 8) // 65536 uint32 = 256KB (all of W as bf16, swizzled)
#define SMEM_BYTES (B_UINTS * 4 + 64 * 4 + 64 * 4) // B region + sMean + sCnt = 262656

// Pack two fp32 -> packed bf16 pair: bias for round-half-up, then one v_perm_b32
// grabbing the high halves (bytes {y3,y2,x3,x2}).
static __device__ __forceinline__ unsigned int pack_bf16(float x, float y) {
    unsigned int xb = __float_as_uint(x) + 0x8000u;
    unsigned int yb = __float_as_uint(y) + 0x8000u;
    return __builtin_amdgcn_perm(yb, xb, 0x07060302u);
}

union AFragU { v16bf v; unsigned int u[8]; };

// One block: 256 threads = 8 waves; each wave -> 16 rows; block -> 128 rows.
__global__ void router_gemm_topk_kernel(const float* __restrict__ hidden,
                                        const float* __restrict__ W,
                                        float* __restrict__ out,
                                        float* __restrict__ gMean,      // [64] fp32 accum
                                        unsigned int* __restrict__ gCnt,// [64]
                                        int totalRows) {
    extern __shared__ char smem[];
    unsigned int* sB      = (unsigned int*)smem;                 // 256KB, bf16-pair swizzled W
    float*        sLogits = (float*)smem;                        // aliases sB after barrier
    float*        sMean   = (float*)(smem + B_UINTS * 4);        // 64 floats
    unsigned int* sCnt    = (unsigned int*)(smem + B_UINTS * 4 + 256);

    const int tid  = threadIdx.x;
    const int lane = tid & 31;
    const int wv   = tid >> 5;

    if (tid < 64) { sMean[tid] = 0.0f; sCnt[tid] = 0u; }

    // ---- Phase 1: stage W (fp32 global) -> bf16 swizzled fragments in LDS ----
    // layout index i: v=i&7 (K-pair), lane=(i>>3)&31, ntile=(i>>8)&3, kIter=i>>10
    for (int i = tid; i < B_UINTS; i += 256) {
        int v     = i & 7;
        int bl    = (i >> 3) & 31;
        int nt    = (i >> 8) & 3;
        int kIt   = i >> 10;
        int e     = nt * 16 + (bl & 15);
        int k     = kIt * 32 + (bl >> 4) * 16 + v * 2;
        const float* wp = W + e * D_DIM + k;
        sB[i] = pack_bf16(wp[0], wp[1]);
    }
    __syncthreads();

    // ---- Phase 2: WMMA GEMM, 16 rows x 64 experts per wave ----
    const int rowBase = blockIdx.x * ROWS_PER_BLOCK + wv * 16;
    const int m  = lane & 15;
    const int hs = lane >> 4;
    const float* rowp = hidden + (size_t)(rowBase + m) * D_DIM;

    v8f acc[4];
    acc[0] = v8f{}; acc[1] = v8f{}; acc[2] = v8f{}; acc[3] = v8f{};

    #pragma unroll 2
    for (int kIt = 0; kIt < KITERS; ++kIt) {
        const int kbase = kIt * 32;
        // Kick off all four B-fragment LDS loads first (2x ds_load_b128 each).
        const unsigned int* bp = &sB[kIt * 1024 + lane * 8];
        const v16bf b0 = *(const v16bf*)(bp);
        const v16bf b1 = *(const v16bf*)(bp + 256);
        const v16bf b2 = *(const v16bf*)(bp + 512);
        const v16bf b3 = *(const v16bf*)(bp + 768);
        // speculative prefetch ~16 iterations ahead of the A stream
        __builtin_prefetch(rowp + kbase + 512, 0, 0);
        // A fragment: group0 = K[kbase+hs*16 .. +7], group1 = K[kbase+hs*16+8 .. +7]
        const float4* g0 = (const float4*)(rowp + kbase + hs * 16);
        const float4* g1 = (const float4*)(rowp + kbase + hs * 16 + 8);
        float4 a0 = g0[0], a1 = g0[1], a2 = g1[0], a3 = g1[1];
        AFragU af;
        af.u[0] = pack_bf16(a0.x, a0.y);
        af.u[1] = pack_bf16(a0.z, a0.w);
        af.u[2] = pack_bf16(a1.x, a1.y);
        af.u[3] = pack_bf16(a1.z, a1.w);
        af.u[4] = pack_bf16(a2.x, a2.y);
        af.u[5] = pack_bf16(a2.z, a2.w);
        af.u[6] = pack_bf16(a3.x, a3.y);
        af.u[7] = pack_bf16(a3.z, a3.w);
        acc[0] = __builtin_amdgcn_wmma_f32_16x16x32_bf16(
            false, af.v, false, b0, (short)0, acc[0], false, false);
        acc[1] = __builtin_amdgcn_wmma_f32_16x16x32_bf16(
            false, af.v, false, b1, (short)0, acc[1], false, false);
        acc[2] = __builtin_amdgcn_wmma_f32_16x16x32_bf16(
            false, af.v, false, b2, (short)0, acc[2], false, false);
        acc[3] = __builtin_amdgcn_wmma_f32_16x16x32_bf16(
            false, af.v, false, b3, (short)0, acc[3], false, false);
    }

    __syncthreads();  // everyone done reading sB; safe to alias with logits

    // C layout: VGPR r, lane L -> row (r + 8*(L>>4)), col (L&15); rows padded to 65
    #pragma unroll
    for (int nt = 0; nt < 4; ++nt) {
        #pragma unroll
        for (int r = 0; r < 8; ++r) {
            sLogits[(wv * 16 + r + 8 * hs) * 65 + nt * 16 + (lane & 15)] = acc[nt][r];
        }
    }
    __syncthreads();

    // ---- Phase 3: per-row softmax + top-8 (threads 0..127 each own one row) ----
    if (tid < ROWS_PER_BLOCK) {
        float* rw = &sLogits[tid * 65];
        float mx = rw[0];
        for (int e = 1; e < 64; ++e) mx = fmaxf(mx, rw[e]);
        float s = 0.0f;
        for (int e = 0; e < 64; ++e) { float p = __expf(rw[e] - mx); rw[e] = p; s += p; }
        float inv = 1.0f / s;
        // probs + staggered mean accumulation (avoid same-address atomic pileup)
        for (int j = 0; j < 64; ++j) {
            int e = (tid + j) & 63;
            float pr = rw[e] * inv;
            rw[e] = pr;
            atomicAdd(&sMean[e], pr);
        }
        // iterated top-8, strict '>' ascending scan => lowest-index tie-break
        int   bi[TOPK];
        float bv[TOPK];
        float sum8 = 0.0f;
        #pragma unroll
        for (int kk = 0; kk < TOPK; ++kk) {
            int best = 0; float bval = -1.0f;
            for (int e = 0; e < 64; ++e) {
                float p = rw[e];
                if (p > bval) { bval = p; best = e; }
            }
            rw[best] = -1.0f;
            bi[kk] = best; bv[kk] = bval; sum8 += bval;
        }
        float invw = 1.0f / (sum8 + 1e-8f);
        const int grow = blockIdx.x * ROWS_PER_BLOCK + tid;
        float* outIdx = out;
        float* outW   = out + (size_t)totalRows * TOPK;
        #pragma unroll
        for (int kk = 0; kk < TOPK; ++kk) {
            outIdx[grow * TOPK + kk] = (float)bi[kk];
            outW[grow * TOPK + kk]   = bv[kk] * invw;
            atomicAdd(&sCnt[bi[kk]], 1u);
        }
    }
    __syncthreads();

    if (tid < 64) {
        atomicAdd(&gMean[tid], sMean[tid]);
        atomicAdd(&gCnt[tid], sCnt[tid]);
    }
}

// Final reduction: KL(uniform || mean_probs)/E and float-cast counts
__global__ void router_finalize_kernel(const float* __restrict__ gMean,
                                       const unsigned int* __restrict__ gCnt,
                                       float* __restrict__ out,
                                       int totalRows) {
    __shared__ float red[64];
    const int e = threadIdx.x;
    const float u = 1.0f / 64.0f;
    float mean = gMean[e] / (float)totalRows;
    red[e] = u * (__logf(u) - __logf(mean));
    __syncthreads();
    const size_t lossOff = (size_t)totalRows * 2 * TOPK;
    if (e == 0) {
        float s = 0.0f;
        for (int i = 0; i < 64; ++i) s += red[i];
        out[lossOff] = s * (1.0f / 64.0f);
    }
    out[lossOff + 1 + e] = (float)gCnt[e];
}

extern "C" void kernel_launch(void* const* d_in, const int* in_sizes, int n_in,
                              void* d_out, int out_size, void* d_ws, size_t ws_size,
                              hipStream_t stream) {
    const float* hidden = (const float*)d_in[0];
    const float* W      = (const float*)d_in[1];
    float* out = (float*)d_out;

    const int totalRows = in_sizes[0] / D_DIM;           // B*S = 16384
    float* gMean       = (float*)d_ws;                   // 64 floats
    unsigned int* gCnt = (unsigned int*)((char*)d_ws + 256);

    (void)hipFuncSetAttribute((const void*)router_gemm_topk_kernel,
                              hipFuncAttributeMaxDynamicSharedMemorySize, SMEM_BYTES);

    hipMemsetAsync(d_ws, 0, 512, stream);

    const int blocks = totalRows / ROWS_PER_BLOCK;       // 128
    router_gemm_topk_kernel<<<blocks, 256, SMEM_BYTES, stream>>>(
        hidden, W, out, gMean, gCnt, totalRows);
    router_finalize_kernel<<<1, 64, 0, stream>>>(gMean, gCnt, out, totalRows);
}
